// TriangleUpdateABG_46823733461368
// MI455X (gfx1250) — matrix operand order repaired and checked by reading, not compiled, not run on hardware
//
#include <hip/hip_runtime.h>
#include <hip/hip_bf16.h>

#define C_Z 128
#define C_H 128
#define EPS_LN 1e-5f
#define ROWS_PER_BLOCK 128
#define THREADS 256
#define ZSTRIDE 136   // ushorts per LDS row (128 + 8 pad to stagger banks)

typedef __attribute__((ext_vector_type(16))) __bf16 v16bf;
typedef __attribute__((ext_vector_type(8)))  __bf16 v8bf;
typedef __attribute__((ext_vector_type(4)))  __bf16 v4bf;
typedef __attribute__((ext_vector_type(8)))  float  v8f;

__device__ __forceinline__ unsigned short f2bf_rne(float f) {
    unsigned u = __float_as_uint(f);
    u += 0x7FFFu + ((u >> 16) & 1u);
    return (unsigned short)(u >> 16);
}

// sigmoid via v_exp_f32 + v_rcp_f32 (avoids the IEEE v_div_scale sequence)
__device__ __forceinline__ float sigmoidf_fast(float v) {
    return __builtin_amdgcn_rcpf(1.0f + __expf(-v));
}

__device__ __forceinline__ v8f splat8(float v) {
    v8f r;
#pragma unroll
    for (int i = 0; i < 8; ++i) r[i] = v;
    return r;
}

// Accumulate one 16x16 output tile: 4 WMMAs over K=128 (4 K-steps of 32).
__device__ __forceinline__ v8f wmma_tile(const v16bf* __restrict__ Bp,
                                         const v16bf a[4], int tile, int lane, v8f acc) {
#pragma unroll
    for (int ks = 0; ks < 4; ++ks) {
        v16bf bf = Bp[(size_t)(tile * 4 + ks) * 32 + lane];
        acc = __builtin_amdgcn_wmma_f32_16x16x32_bf16(
            /*neg_a=*/false, a[ks], /*neg_b=*/false, bf,
            /*c_mod=*/(short)0, acc, /*reuse_a=*/false, /*reuse_b=*/false);
    }
    return acc;
}

// ---------------------------------------------------------------------------
// Kernel 1: pre-swizzle the 5 weight matrices (row-major [h][c], fp32) into
// per-lane bf16 WMMA B-fragments.
//   lane L: n = L&15, half = L>>4; holds K = half*16 + 0..15 linearly.
// Packed layout: fragment id fid = ((m*8 + t)*4 + ks)*32 + lane, 16 bf16 each.
// Total: 40 tiles * 4 ksteps * 32 lanes * 16 bf16 = 160 KB (L2-resident).
// ---------------------------------------------------------------------------
__global__ void pack_weights_kernel(const float* __restrict__ apw,
                                    const float* __restrict__ agw,
                                    const float* __restrict__ bpw,
                                    const float* __restrict__ bgw,
                                    const float* __restrict__ gw,
                                    unsigned short* __restrict__ wpack) {
    int fid = blockIdx.x * blockDim.x + threadIdx.x;
    if (fid >= 40 * 4 * 32) return;
    int lane = fid & 31;
    int rest = fid >> 5;
    int ks   = rest & 3;
    int tile = rest >> 2;          // 0..39
    int m    = tile >> 3;          // 0..4 : a_p, a_g, b_p, b_g, g
    int t    = tile & 7;           // n-tile within matrix
    const float* W = (m == 0) ? apw : (m == 1) ? agw : (m == 2) ? bpw
                   : (m == 3) ? bgw : gw;
    int n  = lane & 15;
    int kh = lane >> 4;
    int ch = t * 16 + n;           // output channel (weight row)
    const float* src = W + (size_t)ch * C_Z + ks * 32 + kh * 16;
    unsigned short* dst = wpack + (size_t)fid * 16;
#pragma unroll
    for (int v = 0; v < 16; ++v) dst[v] = f2bf_rne(src[v]);
}

// ---------------------------------------------------------------------------
// Kernel 2: fused LayerNorm + 5 projections + gating.
// Block: 256 threads (8 wave32), 128 rows. Each wave owns 16 rows.
// ---------------------------------------------------------------------------
__global__ __launch_bounds__(THREADS)
void triangle_fused_kernel(const float* __restrict__ x,
                           const float* __restrict__ mask,
                           const float* __restrict__ lnw,
                           const float* __restrict__ lnb,
                           const float* __restrict__ apb,
                           const float* __restrict__ agb,
                           const float* __restrict__ bpb,
                           const float* __restrict__ bgb,
                           const float* __restrict__ gbias,
                           const unsigned short* __restrict__ wpack,
                           float* __restrict__ outa,
                           float* __restrict__ outb,
                           float* __restrict__ outg) {
    __shared__ unsigned short zlds[ROWS_PER_BLOCK * ZSTRIDE];  // bf16 z rows
    __shared__ float mlds[ROWS_PER_BLOCK];
    __shared__ float wlds[C_Z];
    __shared__ float blds[C_Z];

    const int tid = threadIdx.x;
    const int rowbase = blockIdx.x * ROWS_PER_BLOCK;

    if (tid < C_Z) { wlds[tid] = lnw[tid]; blds[tid] = lnb[tid]; }
    if (tid < ROWS_PER_BLOCK) mlds[tid] = mask[rowbase + tid];
    __syncthreads();

    // ---------------- LayerNorm: 2 threads per row, 64 floats each ----------
    {
        const int rl   = tid >> 1;     // local row 0..127
        const int half = tid & 1;
        const float4* xp = (const float4*)(x + (size_t)(rowbase + rl) * C_Z + half * 64);
        float4 v[16];
        float s = 0.f, sq = 0.f;
#pragma unroll
        for (int i = 0; i < 16; ++i) {
            v[i] = xp[i];
            s  += v[i].x + v[i].y + v[i].z + v[i].w;
            sq += v[i].x * v[i].x + v[i].y * v[i].y + v[i].z * v[i].z + v[i].w * v[i].w;
        }
        // partner thread (lane^1, same wave) holds the other half of the row
        s  += __shfl_xor(s, 1, 32);
        sq += __shfl_xor(sq, 1, 32);
        const float mu   = s * (1.0f / C_Z);
        const float rstd = rsqrtf(sq * (1.0f / C_Z) - mu * mu + EPS_LN);

        unsigned short* zp = &zlds[rl * ZSTRIDE + half * 64];
#pragma unroll
        for (int i = 0; i < 16; ++i) {
            const int c = half * 64 + i * 4;
            const float4 wv4 = *(const float4*)&wlds[c];
            const float4 bv4 = *(const float4*)&blds[c];
            // pairs through __bf16 vector -> v_cvt_pk_bf16_f32
            v4bf pk;
            pk[0] = (__bf16)((v[i].x - mu) * rstd * wv4.x + bv4.x);
            pk[1] = (__bf16)((v[i].y - mu) * rstd * wv4.y + bv4.y);
            pk[2] = (__bf16)((v[i].z - mu) * rstd * wv4.z + bv4.z);
            pk[3] = (__bf16)((v[i].w - mu) * rstd * wv4.w + bv4.w);
            *(v4bf*)&zp[i * 4] = pk;
        }
    }
    __syncthreads();

    // ---------------- WMMA phase: each wave computes 16 rows x 640 cols -----
    const int lane = tid & 31;
    const int wv   = tid >> 5;
    const int n    = lane & 15;            // column within 16-wide tile
    const int koff = (lane >> 4) * 8;      // 0 for lanes 0-15, 8 for 16-31

    // Build A fragments (16x32 bf16 each, 4 K-steps).
    // A layout (matches packed B): lanes 0-15 hold M=lane, K = 0..15;
    // lanes 16-31 hold M=lane-16, K = 16..31.
    v16bf afrag[4];
    {
        const unsigned short* zrow = &zlds[(wv * 16 + n) * ZSTRIDE];
#pragma unroll
        for (int ks = 0; ks < 4; ++ks) {
            v8bf lo = *(const v8bf*)&zrow[ks * 32 + 2 * koff];
            v8bf hi = *(const v8bf*)&zrow[ks * 32 + 2 * koff + 8];
            afrag[ks] = __builtin_shufflevector(lo, hi,
                0, 1, 2, 3, 4, 5, 6, 7, 8, 9, 10, 11, 12, 13, 14, 15);
        }
    }

    // Mask values for this lane's 8 output rows (C/D layout: M = r + koff).
    float mrow[8];
    {
        const float* mp = &mlds[wv * 16 + koff];
#pragma unroll
        for (int r = 0; r < 8; ++r) mrow[r] = mp[r];
    }

    const v16bf* Bp = (const v16bf*)wpack;
    const size_t orow = (size_t)(rowbase + wv * 16 + koff) * C_H;

    // ---- a = mask * sigmoid(a_g) * a_p ----
    for (int t = 0; t < 8; ++t) {
        v8f accp = wmma_tile(Bp, afrag, 0 * 8 + t, lane, splat8(apb[t * 16 + n]));
        v8f accg = wmma_tile(Bp, afrag, 1 * 8 + t, lane, splat8(agb[t * 16 + n]));
        float* op = outa + orow + t * 16 + n;
#pragma unroll
        for (int r = 0; r < 8; ++r)
            op[(size_t)r * C_H] = mrow[r] * sigmoidf_fast(accg[r]) * accp[r];
    }

    // ---- b = mask * sigmoid(b_g) * b_p ----
    for (int t = 0; t < 8; ++t) {
        v8f accp = wmma_tile(Bp, afrag, 2 * 8 + t, lane, splat8(bpb[t * 16 + n]));
        v8f accg = wmma_tile(Bp, afrag, 3 * 8 + t, lane, splat8(bgb[t * 16 + n]));
        float* op = outb + orow + t * 16 + n;
#pragma unroll
        for (int r = 0; r < 8; ++r)
            op[(size_t)r * C_H] = mrow[r] * sigmoidf_fast(accg[r]) * accp[r];
    }

    // ---- g = sigmoid(Linear_g(z)) ----
    for (int t = 0; t < 8; ++t) {
        v8f accg = wmma_tile(Bp, afrag, 4 * 8 + t, lane, splat8(gbias[t * 16 + n]));
        float* op = outg + orow + t * 16 + n;
#pragma unroll
        for (int r = 0; r < 8; ++r)
            op[(size_t)r * C_H] = sigmoidf_fast(accg[r]);
    }
}

extern "C" void kernel_launch(void* const* d_in, const int* in_sizes, int n_in,
                              void* d_out, int out_size, void* d_ws, size_t ws_size,
                              hipStream_t stream) {
    const float* x      = (const float*)d_in[0];   // [1,512,512,128]
    const float* mask   = (const float*)d_in[1];   // [1,512,512,1]
    const float* lnw    = (const float*)d_in[2];
    const float* lnb    = (const float*)d_in[3];
    const float* apw    = (const float*)d_in[4];
    const float* apb    = (const float*)d_in[5];
    const float* agw    = (const float*)d_in[6];
    const float* agb    = (const float*)d_in[7];
    const float* bpw    = (const float*)d_in[8];
    const float* bpb    = (const float*)d_in[9];
    const float* bgw    = (const float*)d_in[10];
    const float* bgb    = (const float*)d_in[11];
    const float* gw     = (const float*)d_in[12];
    const float* gb     = (const float*)d_in[13];
    (void)n_in; (void)out_size; (void)ws_size;

    const int rows = in_sizes[1];                  // N*N = 262144
    unsigned short* wpack = (unsigned short*)d_ws; // 160 KB packed bf16 weights

    pack_weights_kernel<<<(40 * 4 * 32 + 255) / 256, 256, 0, stream>>>(
        apw, agw, bpw, bgw, gw, wpack);

    float* outa = (float*)d_out;
    float* outb = outa + (size_t)rows * C_H;
    float* outg = outb + (size_t)rows * C_H;

    triangle_fused_kernel<<<rows / ROWS_PER_BLOCK, THREADS, 0, stream>>>(
        x, mask, lnw, lnb, apb, agb, bpb, bgb, gb, wpack, outa, outb, outg);
}
